// Convolution_61993557950960
// MI455X (gfx1250) — matrix operand order repaired and checked by reading, not compiled, not run on hardware
//
#include <hip/hip_runtime.h>
#include <hip/hip_bf16.h>
#include <stdint.h>

// ---------------- CDNA5 WMMA vector types ----------------
typedef __attribute__((ext_vector_type(16))) __bf16 v16bf;
typedef __attribute__((ext_vector_type(8)))  float  v8f;
typedef int v4i_vs __attribute__((vector_size(16)));   // matches async-LDS builtin param

union BFrag16 { v16bf v; uint32_t u[8]; };

#define NSP   64000   // 40^3 voxels
#define DIMW  40
#define CIN   64
#define COUT  144
#define NOFF  125     // 5^3 kernel taps
#define A_ELEMS   (125*2*9*32*16)  // bf16 elements of swizzled A (1,152,000)
#define A_O_ELEMS (2*9*32*16)      // bf16 elements per tap (9216) = 18432 bytes

// CDNA5 async global->LDS path (guarded: falls back to sync LDS copy)
#if __has_builtin(__builtin_amdgcn_global_load_async_to_lds_b128)
#define USE_ASYNC 1
#else
#define USE_ASYNC 0
#endif

__device__ __forceinline__ void wait_async0() {
#if __has_builtin(__builtin_amdgcn_s_wait_asynccnt)
  __builtin_amdgcn_s_wait_asynccnt(0);
#else
  asm volatile("s_wait_asynccnt 0" ::: "memory");
#endif
}

__device__ __forceinline__ void zero_frag(BFrag16& f) {
  #pragma unroll
  for (int q = 0; q < 8; ++q) f.u[q] = 0u;
}

__device__ __constant__ float c_C112[3][3][5] = {
  { { 0.f, 0.f, -0.18257418583505536f, 0.f, -0.31622776601683794f },
    { 0.f, 0.31622776601683794f, 0.f, 0.f, 0.f },
    { 0.31622776601683794f, 0.f, 0.f, 0.f, 0.f } },
  { { 0.f, 0.31622776601683794f, 0.f, 0.f, 0.f },
    { 0.f, 0.f, 0.3651483716701107f, 0.f, 0.f },
    { 0.f, 0.f, 0.f, 0.31622776601683794f, 0.f } },
  { { 0.31622776601683794f, 0.f, 0.f, 0.f, 0.f },
    { 0.f, 0.f, 0.f, 0.31622776601683794f, 0.f },
    { 0.f, 0.f, -0.18257418583505536f, 0.f, 0.31622776601683794f } }
};

__device__ __forceinline__ unsigned short f32_to_bf16(float f) {
  union { float f; uint32_t u; } v; v.f = f;
  uint32_t r = v.u + 0x7FFFu + ((v.u >> 16) & 1u);   // round-to-nearest-even
  return (unsigned short)(r >> 16);
}

__device__ __forceinline__ float sus_f(float t) {
  return (t > 0.f) ? __expf(-1.f / t) : 0.f;
}

// ============================================================
// Kernel 1: build conv kernel, pre-swizzled to WMMA A-layout, bf16, *0.1 folded
//   Aswz[(((o*2+chunk)*9 + t)*32 + lane)*16 + e]
//   A 16x32 layout (contiguous K, matches B): K = 16*g + e, lane = m + 16*g
// ============================================================
__global__ void build_kernelA(const float* __restrict__ w_tp,
                              unsigned short* __restrict__ Aswz) {
  int idx = blockIdx.x * 256 + threadIdx.x;          // 125*144*64 = 1,152,000
  if (idx >= NOFF * COUT * CIN) return;
  int ci = idx & 63;
  int co = (idx >> 6) % COUT;
  int o  = idx / (COUT * CIN);
  int kz = o % 5, ky = (o / 5) % 5, kx = o / 25;

  float rx = 0.5f * (float)(kx - 2);
  float ry = 0.5f * (float)(ky - 2);
  float rz = 0.5f * (float)(kz - 2);
  float d  = sqrtf(rx*rx + ry*ry + rz*rz);

  float emb[5];
  const float amp = 1.14136f * __expf(2.0f);
  #pragma unroll
  for (int r = 0; r < 5; ++r) {
    float vr   = (float)(r + 1) * (1.0f / 6.0f);
    float diff = (d - vr) * 6.0f;
    emb[r] = amp * sus_f(diff + 1.0f) * sus_f(1.0f - diff);
  }

  float dn = fmaxf(d, 1e-12f);
  float ux = rx / dn, uy = ry / dn, uz = rz / dn;
  const float SQ3 = 1.7320508075688772f;
  const float SQ5 = 2.2360679774997896f;
  float s1[3] = { SQ3 * ux, SQ3 * uy, SQ3 * uz };
  float s2[5] = { SQ5 * SQ3 * ux * uz,
                  SQ5 * SQ3 * ux * uy,
                  SQ5 * (uy*uy - 0.5f * (ux*ux + uz*uz)),
                  SQ5 * SQ3 * uy * uz,
                  SQ5 * 0.5f * SQ3 * (uz*uz - ux*ux) };

  const float invS15 = 0.08944271909999159f;          // 1/5^1.5
  auto Wv = [&](int wi, int u, int w) -> float {
    const float* p = w_tp + wi * 256 + u * 16 + w;    // w_tp (5, 1792) row-major
    float s = 0.f;
    #pragma unroll
    for (int r = 0; r < 5; ++r) s += emb[r] * p[r * 1792];
    return s * invS15;
  };

  const float PW0 = 0.1767766952966369f;              // sqrt(1/32)
  const float PW1 = 0.25f;                            // sqrt(3/48)
  const float PW2 = 0.3952847075210474f;              // sqrt(5/32)

  float val;
  if (ci < 16) {
    int u = ci;
    if (co < 16) {
      val = PW0 * Wv(0, u, co);                                   // B00
    } else if (co < 64) {
      int jc = co - 16;                                           // B01: col = 3w+k
      val = (PW1 / SQ3) * s1[jc % 3] * Wv(1, u, jc / 3);
    } else {
      int jc = co - 64;                                           // B02: col = 5w+k
      val = (PW2 / SQ5) * s2[jc % 5] * Wv(2, u, jc / 5);
    }
  } else {
    int j = ci - 16, u = j / 3, i = j % 3;
    if (co < 16) {
      val = (PW0 / SQ3) * s1[i] * Wv(4, u, co);                   // B10
    } else if (co < 64) {
      int jc = co - 16, w = jc / 3, k = jc % 3;                   // B11
      float cA = 0.f;
      #pragma unroll
      for (int jj = 0; jj < 5; ++jj) cA += c_C112[i][k][jj] * s2[jj]; // C121[i,j,k]=C112[i,k,j]
      val = ((i == k) ? (PW1 / SQ3) * Wv(3, u, w) : 0.f) + PW1 * cA * Wv(6, u, w);
    } else {
      int jc = co - 64, w = jc / 5, k = jc % 5;                   // B12
      float cB = 0.f;
      #pragma unroll
      for (int jj = 0; jj < 3; ++jj) cB += c_C112[i][jj][k] * s1[jj];
      val = PW2 * cB * Wv(5, u, w);
    }
  }
  val *= 0.1f;                                        // fold conv scale

  int chunk = ci >> 5;
  int kk    = ci & 31;
  int gg    = kk >> 4;
  int e     = kk & 15;
  int lane  = (co & 15) + (gg << 4);
  int t     = co >> 4;
  size_t aidx = ((((size_t)o * 2 + chunk) * 9 + t) * 32 + lane) * 16 + e;
  Aswz[aidx] = f32_to_bf16(val);
}

// ============================================================
// Kernel 2: NCDHW f32 -> channels-last bf16  x_t[b][voxel][cin]
// ============================================================
__global__ void transpose_x(const float* __restrict__ x,
                            unsigned short* __restrict__ xt) {
  __shared__ unsigned short tile[64 * 64];
  int blk = blockIdx.x;                  // 2000 blocks: b*1000 + vtile
  int b   = blk / 1000;
  int v0  = (blk % 1000) * 64;
  for (int idx = threadIdx.x; idx < 4096; idx += 256) {
    int ci = idx >> 6, j = idx & 63;     // coalesced read along voxels
    tile[j * 64 + ci] = f32_to_bf16(x[((size_t)(b * CIN + ci)) * NSP + v0 + j]);
  }
  __syncthreads();
  size_t base = ((size_t)b * NSP + v0) * CIN;
  for (int idx = threadIdx.x; idx < 4096; idx += 256) {
    xt[base + idx] = tile[idx];          // coalesced write along channels
  }
}

// ============================================================
// Kernel 3: implicit-GEMM conv via v_wmma_f32_16x16x32_bf16
//  - block = 4 waves; each wave owns 32 voxels (two 16x16 N-tiles)
//  - per-tap A block (18.4KB) staged global->LDS with async DMA,
//    double-buffered, shared by all 4 waves
// ============================================================
__global__ void __launch_bounds__(128)
conv_wmma(const unsigned short* __restrict__ xt,
          const unsigned short* __restrict__ Asw,
          float* __restrict__ out) {
  __shared__ __align__(16) unsigned short Alds[2][A_O_ELEMS];

  const int tid  = threadIdx.x;
  const int wave = tid >> 5;
  const int lane = tid & 31;
  const int tile = blockIdx.x * 4 + wave;      // 0..3999
  const int b     = tile / 2000;
  const int vbase = (tile % 2000) * 32;

  const int n = lane & 15;
  const int g = lane >> 4;

  int vox0 = vbase + n;
  int vox1 = vbase + 16 + n;
  int vz0 = vox0 % DIMW, vy0 = (vox0 / DIMW) % DIMW, vx0 = vox0 / (DIMW * DIMW);
  int vz1 = vox1 % DIMW, vy1 = (vox1 / DIMW) % DIMW, vx1 = vox1 / (DIMW * DIMW);

  v8f acc0[9], acc1[9];
  #pragma unroll
  for (int t = 0; t < 9; ++t) {
    #pragma unroll
    for (int r = 0; r < 8; ++r) { acc0[t][r] = 0.f; acc1[t][r] = 0.f; }
  }

  const unsigned short* xb = xt + (size_t)b * NSP * CIN;

  // stage one tap's A block (18432 B) into LDS buffer `buf`
  auto stage = [&](int o, int buf) {
    const char* src = (const char*)(Asw + (size_t)o * A_O_ELEMS);
    char* dst = (char*)&Alds[buf][0];
#if USE_ASYNC
    #pragma unroll
    for (int q = 0; q < 9; ++q) {
      int off = (q * 128 + tid) * 16;
      __builtin_amdgcn_global_load_async_to_lds_b128(
          (__attribute__((address_space(1))) v4i_vs*)(src + off),
          (__attribute__((address_space(3))) v4i_vs*)(dst + off),
          0, 0);
    }
#else
    #pragma unroll
    for (int q = 0; q < 9; ++q) {
      int off = (q * 128 + tid) * 16;
      *(v4i_vs*)(dst + off) = *(const v4i_vs*)(src + off);
    }
#endif
  };

  stage(0, 0);

  int o = 0;
  for (int kx = 0; kx < 5; ++kx) {
    int nx0 = vx0 + kx - 2, nx1 = vx1 + kx - 2;
    bool okx0 = (unsigned)nx0 < (unsigned)DIMW;
    bool okx1 = (unsigned)nx1 < (unsigned)DIMW;
    for (int ky = 0; ky < 5; ++ky) {
      int ny0 = vy0 + ky - 2, ny1 = vy1 + ky - 2;
      bool oky0 = okx0 && ((unsigned)ny0 < (unsigned)DIMW);
      bool oky1 = okx1 && ((unsigned)ny1 < (unsigned)DIMW);
      for (int kz = 0; kz < 5; ++kz, ++o) {
        int nz0 = vz0 + kz - 2, nz1 = vz1 + kz - 2;
        bool ok0 = oky0 && ((unsigned)nz0 < (unsigned)DIMW);
        bool ok1 = oky1 && ((unsigned)nz1 < (unsigned)DIMW);
        const int cur = o & 1;

#if USE_ASYNC
        wait_async0();                 // own async copies done
#endif
        __syncthreads();               // all waves' copies visible
        if (o + 1 < NOFF) stage(o + 1, cur ^ 1);

        int cx0 = ok0 ? nx0 : 0, cy0 = ok0 ? ny0 : 0, cz0 = ok0 ? nz0 : 0;
        int cx1 = ok1 ? nx1 : 0, cy1 = ok1 ? ny1 : 0, cz1 = ok1 ? nz1 : 0;
        size_t nv0 = ((size_t)cx0 * DIMW + cy0) * DIMW + cz0;
        size_t nv1 = ((size_t)cx1 * DIMW + cy1) * DIMW + cz1;
        const unsigned short* bp0 = xb + nv0 * CIN + g * 16;
        const unsigned short* bp1 = xb + nv1 * CIN + g * 16;

        #pragma unroll
        for (int c = 0; c < 2; ++c) {              // K chunks of 32 cin
          BFrag16 b0, b1;
          if (ok0) b0.v = *(const v16bf*)(bp0 + c * 32);
          else     zero_frag(b0);
          if (ok1) b1.v = *(const v16bf*)(bp1 + c * 32);
          else     zero_frag(b1);
          const unsigned short* ap = &Alds[cur][(c * 9) * 512 + lane * 16];
          #pragma unroll
          for (int t = 0; t < 9; ++t) {
            v16bf av = *(const v16bf*)(ap + t * 512);
            acc0[t] = __builtin_amdgcn_wmma_f32_16x16x32_bf16(
                false, av, false, b0.v, (short)0, acc0[t], false, false);
            acc1[t] = __builtin_amdgcn_wmma_f32_16x16x32_bf16(
                false, av, false, b1.v, (short)0, acc1[t], false, false);
          }
        }
        if (o + 1 < NOFF) __syncthreads();   // buffer `cur` free for tap o+2
      }
    }
  }

  // store D: element (m = r + 8*g, n) -> out[b, 16*t+m, vbase(+16)+n]
  float* ob = out + (size_t)b * COUT * NSP + vbase + n;
  #pragma unroll
  for (int t = 0; t < 9; ++t) {
    #pragma unroll
    for (int r = 0; r < 8; ++r) {
      ob[(size_t)(16 * t + r + 8 * g) * NSP]      = acc0[t][r];
      ob[(size_t)(16 * t + r + 8 * g) * NSP + 16] = acc1[t][r];
    }
  }
}

// ============================================================
// Kernel 4: add self-interaction term into d_out (channels 0..63)
// ============================================================
__global__ void sc_add(const float* __restrict__ x,
                       const float* __restrict__ w0,
                       const float* __restrict__ w1,
                       float* __restrict__ out) {
  __shared__ float ws0[256];
  __shared__ float ws1[256];
  int tid = threadIdx.x;                 // 256 threads
  ws0[tid] = w0[tid];
  ws1[tid] = w1[tid];
  __syncthreads();

  int idx = blockIdx.x * 256 + tid;      // 0..127999 == 2*64000
  int b = idx / NSP;
  int v = idx % NSP;

  float xv[64];
  #pragma unroll
  for (int c = 0; c < 64; ++c)
    xv[c] = x[((size_t)(b * CIN + c)) * NSP + v];

  float* o = out + (size_t)b * COUT * NSP + v;
  #pragma unroll
  for (int w = 0; w < 16; ++w) {
    float s = 0.f;
    #pragma unroll
    for (int u = 0; u < 16; ++u) s += ws0[u * 16 + w] * xv[u];
    o[(size_t)w * NSP] += 0.25f * s;
  }
  #pragma unroll
  for (int w = 0; w < 16; ++w) {
    #pragma unroll
    for (int i = 0; i < 3; ++i) {
      float s = 0.f;
      #pragma unroll
      for (int u = 0; u < 16; ++u) s += ws1[u * 16 + w] * xv[16 + 3 * u + i];
      o[(size_t)(16 + 3 * w + i) * NSP] += 0.25f * s;
    }
  }
}

// ============================================================
extern "C" void kernel_launch(void* const* d_in, const int* in_sizes, int n_in,
                              void* d_out, int out_size, void* d_ws, size_t ws_size,
                              hipStream_t stream) {
  const float* x     = (const float*)d_in[0];   // (2,64,40,40,40)
  const float* w_sc0 = (const float*)d_in[1];   // (16,16)
  const float* w_sc1 = (const float*)d_in[2];   // (16,16)
  const float* w_tp  = (const float*)d_in[3];   // (5,1792)
  float* out = (float*)d_out;                   // (2,144,40,40,40)

  unsigned short* Aswz = (unsigned short*)d_ws;
  unsigned short* xt   = (unsigned short*)((char*)d_ws + (size_t)A_ELEMS * 2);

  build_kernelA<<<4500, 256, 0, stream>>>(w_tp, Aswz);   // 125*144*64 threads
  transpose_x  <<<2000, 256, 0, stream>>>(x, xt);        // 2*1000 voxel tiles
  conv_wmma    <<<1000, 128, 0, stream>>>(xt, Aswz, out);// 4000 waves, 32 voxels each
  sc_add       <<< 500, 256, 0, stream>>>(x, w_sc0, w_sc1, out);
}
